// GcnNet_91087666414240
// MI455X (gfx1250) — matrix operand order repaired and checked by reading, not compiled, not run on hardware
//
#include <hip/hip_runtime.h>
#include <hip/hip_bf16.h>

#define NN     20000
#define EE     400000
#define HH     256
#define BAG    50
#define LBL    3000
#define NGCN   2

typedef __attribute__((ext_vector_type(16))) __bf16 v16bf;
typedef __attribute__((ext_vector_type(8)))  float  v8f;
typedef __attribute__((ext_vector_type(4)))  unsigned int u32x4;
typedef __attribute__((ext_vector_type(8)))  int  i32x8;
typedef __attribute__((ext_vector_type(4)))  int  i32x4;

// ---------------------------------------------------------------------------
// TDM helper: async 2D tile DMA global -> LDS via Tensor Data Mover.
// D# per CDNA5 ISA:
//   group0: count=1 | lds_addr | global_addr[56:0] | type=2
//   group1: data_size=4B, tensor_dim0/1 = remaining extent from tile start
//           (OOB elements return ZERO -> free edge padding), tile dims, pitch.
// Issued by one wave; completion tracked with TENSORcnt (in order per wave).
// This toolchain exposes the 6-arg builtin form (clang-23 / therock-10.0).
// ---------------------------------------------------------------------------
__device__ __forceinline__ unsigned lds_off_of(const void* p) {
    // LDS aperture flat address: low 32 bits are the LDS byte offset
    return (unsigned)(unsigned long long)(uintptr_t)p;
}

__device__ __forceinline__ void tdm_load_2d(unsigned lds_off,
                                            const float* gptr,
                                            unsigned tdim0, unsigned tdim1,
                                            unsigned stride0,
                                            unsigned tile0, unsigned tile1) {
    const unsigned long long ga = (unsigned long long)(uintptr_t)gptr;
    u32x4 g0 = {
        1u,                                                 // count=1 (valid D#)
        lds_off,                                            // lds_addr
        (unsigned)ga,                                       // global_addr[31:0]
        (unsigned)((ga >> 32) & 0x01FFFFFFu) | (2u << 30)   // addr[56:32] | type=2
    };
    i32x8 g1 = {
        (int)(2u << 16),                                    // data_size = 4 bytes
        (int)((tdim0 & 0xFFFFu) << 16),                     // tensor_dim0[15:0]
        (int)((tdim0 >> 16) | ((tdim1 & 0xFFFFu) << 16)),   // td0[31:16] | td1[15:0]
        (int)(((tdim1 >> 16) & 0xFFFFu) | (tile0 << 16)),   // td1[31:16] | tile_dim0
        (int)(tile1 & 0xFFFFu),                             // tile_dim1 (tile_dim2=0)
        (int)stride0,                                       // tensor_dim0_stride[31:0]
        0, 0
    };
    i32x4 z4 = {};
    i32x8 z8 = {};
    __builtin_amdgcn_tensor_load_to_lds(g0, g1, z4, z4, z8, 0);
}

// ---------------------------------------------------------------------------
// 1) EmbeddingBag(sum, per-sample weights) + bias + relu  ->  h [N,H]
// ---------------------------------------------------------------------------
__global__ void embed_kernel(const int* __restrict__ feat_idx,
                             const float* __restrict__ feat_w,
                             const float* __restrict__ W_in,
                             const float* __restrict__ b_in,
                             float* __restrict__ h) {
    __shared__ int   sidx[BAG];
    __shared__ float sw[BAG];
    const int nid = blockIdx.x;
    const int c   = threadIdx.x;
    if (c < BAG) {
        sidx[c] = feat_idx[nid * BAG + c];
        sw[c]   = feat_w[nid * BAG + c];
    }
    __syncthreads();
    float acc = b_in[c];
    #pragma unroll 5
    for (int i = 0; i < BAG; ++i)
        acc += W_in[sidx[i] * HH + c] * sw[i];
    h[nid * HH + c] = fmaxf(acc, 0.0f);
}

// ---------------------------------------------------------------------------
// 2) Edge scatter: one block per edge; wave-uniform edge scalars; two
//    relaxed agent-scope f32 atomics (global_atomic_add_f32) into L2.
// ---------------------------------------------------------------------------
__global__ void scatter_kernel(const float* __restrict__ h,
                               const int* __restrict__ src,
                               const int* __restrict__ dst,
                               const float* __restrict__ ppi,
                               const float* __restrict__ selfw,
                               float* __restrict__ ppi_out,
                               float* __restrict__ res) {
    const int e = blockIdx.x;
    const int c = threadIdx.x;
    const int s = src[e];
    const int d = dst[e];
    const float pe = ppi[e];
    const float we = selfw[e];
    const float m  = h[s * HH + c];
    __hip_atomic_fetch_add(ppi_out + d * HH + c, m * pe,
                           __ATOMIC_RELAXED, __HIP_MEMORY_SCOPE_AGENT);
    __hip_atomic_fetch_add(res + d * HH + c, m * we,
                           __ATOMIC_RELAXED, __HIP_MEMORY_SCOPE_AGENT);
}

// ---------------------------------------------------------------------------
// 3) GEMM  C[M,Ncols] = act( A[M,256] @ B[256,ldb] + bias (+ res) )
//    bf16 WMMA 16x16x32, f32 accumulate. Block = 8 waves = 128x16 out tile.
//    A (128x32) and B (32x16) tiles DMA'd into double-buffered LDS by the
//    TDM (wave 0 issues, s_wait_tensorcnt + barrier publishes), 2-deep
//    software pipeline over the 8 K-steps.
// ---------------------------------------------------------------------------
template <bool RELU, bool HAS_RES>
__global__ __launch_bounds__(256)
void gemm_wmma_kernel(const float* __restrict__ A,
                      const float* __restrict__ B, int ldb,
                      const float* __restrict__ bias,
                      const float* __restrict__ res, int ldres,
                      float* __restrict__ C, int ldc,
                      int M, int Ncols) {
    __shared__ float lsA[2][128 * 32];   // 2 x 16 KB
    __shared__ float lsB[2][32 * 16];    // 2 x  2 KB

    const int tid     = threadIdx.x;
    const int wave    = tid >> 5;
    const int lane    = tid & 31;
    const int rowBase = blockIdx.x * 128;
    const int colBase = blockIdx.y * 16;

    const int mrow  = lane & 15;      // row (A/C) or col (B) within 16x16 tile
    const int khalf = lane >> 4;      // K half-group owned by this lane

    const unsigned rowsLeft = (unsigned)(M - rowBase);       // A zero-fill edge
    const unsigned colsLeft = (unsigned)(Ncols - colBase);   // B zero-fill edge

    // prologue: DMA K-step 0 into buffer 0
    if (tid == 0) {
        tdm_load_2d(lds_off_of(&lsA[0][0]), A + (size_t)rowBase * HH,
                    /*tdim0=*/32, /*tdim1=*/rowsLeft, /*stride0=*/HH,
                    /*tile0=*/32, /*tile1=*/128);
        tdm_load_2d(lds_off_of(&lsB[0][0]), B + colBase,
                    /*tdim0=*/colsLeft, /*tdim1=*/32, /*stride0=*/(unsigned)ldb,
                    /*tile0=*/16, /*tile1=*/32);
    }

    v8f acc = {};
    int buf = 0;

    #pragma unroll
    for (int it = 0; it < HH / 32; ++it) {
        if (tid == 0) {
            if (it + 1 < HH / 32) {
                const int k0n = (it + 1) * 32;
                tdm_load_2d(lds_off_of(&lsA[buf ^ 1][0]),
                            A + (size_t)rowBase * HH + k0n,
                            32, rowsLeft, HH, 32, 128);
                tdm_load_2d(lds_off_of(&lsB[buf ^ 1][0]),
                            B + (size_t)k0n * ldb + colBase,
                            colsLeft, 32, (unsigned)ldb, 16, 32);
                __builtin_amdgcn_s_wait_tensorcnt(2);  // current pair landed
            } else {
                __builtin_amdgcn_s_wait_tensorcnt(0);  // last pair landed
            }
        }
        __syncthreads();

        const float* As = &lsA[buf][0];
        const float* Bs = &lsB[buf][0];

        // A fragment: lane m=lane%16; VGPR0-3: K = khalf*8 + 0..7,
        //             VGPR4-7: K = 16 + khalf*8 + 0..7
        v16bf afrag;
        const int arow = (16 * wave + mrow) * 32;
        #pragma unroll
        for (int j = 0; j < 16; ++j) {
            const int v = j >> 1;
            const int k = ((v < 4) ? 0 : 16) + khalf * 8 + (v & 3) * 2 + (j & 1);
            afrag[j] = (__bf16)As[arow + k];
        }
        // B fragment: lane n=lane%16; j<8: K = khalf*8 + j, j>=8: K = 16 + khalf*8 + (j-8)
        v16bf bfrag;
        #pragma unroll
        for (int j = 0; j < 16; ++j)
            bfrag[j] = (__bf16)Bs[(((j < 8) ? 0 : 16) + khalf * 8 + (j & 7)) * 16 + mrow];

        acc = __builtin_amdgcn_wmma_f32_16x16x32_bf16(
            false, afrag, false, bfrag, (short)0, acc, false, false);

        __syncthreads();   // all waves done with buf before TDM rewrites it
        buf ^= 1;
    }

    // epilogue: C VGPR v holds (m = v + 8*khalf, n = lane%16)
    const int col = colBase + mrow;
    if (col < Ncols) {
        const float bv = bias[col];
        #pragma unroll
        for (int v = 0; v < 8; ++v) {
            const int m = rowBase + 16 * wave + v + 8 * khalf;
            if (m < M) {
                float val = acc[v] + bv;
                if (HAS_RES) val += res[m * ldres + col];
                if (RELU)    val = fmaxf(val, 0.0f);
                C[m * ldc + col] = val;
            }
        }
    }
}

// ---------------------------------------------------------------------------
extern "C" void kernel_launch(void* const* d_in, const int* in_sizes, int n_in,
                              void* d_out, int out_size, void* d_ws, size_t ws_size,
                              hipStream_t stream) {
    const int*   feat_idx = (const int*)  d_in[0];
    // d_in[1] = offsets: uniform (i*BAG) -> segment id == i/BAG, not needed
    const float* feat_w   = (const float*)d_in[2];
    const int*   src      = (const int*)  d_in[3];
    const int*   dst      = (const int*)  d_in[4];
    const float* ppi      = (const float*)d_in[5];
    const float* selfw    = (const float*)d_in[6];
    const float* W_in     = (const float*)d_in[7];
    const float* b_in     = (const float*)d_in[8];
    const float* W_upd    = (const float*)d_in[9];
    const float* b_upd    = (const float*)d_in[10];
    const float* W_out    = (const float*)d_in[11];
    const float* b_out    = (const float*)d_in[12];
    float* out = (float*)d_out;

    float* h       = (float*)d_ws;          // N*H
    float* ppi_out = h + (size_t)NN * HH;   // N*H
    float* res     = ppi_out + (size_t)NN * HH;

    embed_kernel<<<NN, HH, 0, stream>>>(feat_idx, feat_w, W_in, b_in, h);

    const int mtiles = (NN + 127) / 128;

    for (int l = 0; l < NGCN; ++l) {
        (void)hipMemsetAsync(ppi_out, 0, (size_t)NN * HH * sizeof(float), stream);
        (void)hipMemsetAsync(res,     0, (size_t)NN * HH * sizeof(float), stream);

        scatter_kernel<<<EE, HH, 0, stream>>>(
            h, src + (size_t)l * EE, dst + (size_t)l * EE,
            ppi + (size_t)l * EE, selfw + (size_t)l * EE, ppi_out, res);

        gemm_wmma_kernel<true, true><<<dim3(mtiles, HH / 16), 256, 0, stream>>>(
            ppi_out, W_upd + (size_t)l * HH * HH, HH,
            b_upd + (size_t)l * HH, res, HH, h, HH, NN, HH);
    }

    gemm_wmma_kernel<false, false><<<dim3(mtiles, (LBL + 15) / 16), 256, 0, stream>>>(
        h, W_out, LBL, b_out, nullptr, 0, out, LBL, NN, LBL);
}
